// VotePoolLayer_52166672778033
// MI455X (gfx1250) — matrix operand order repaired and checked by reading, not compiled, not run on hardware
//
#include <hip/hip_runtime.h>
#include <hip/hip_bf16.h>

// ---------------------------------------------------------------------------
// VotePool layer for MI455X (gfx1250, wave32).
//
// Roofline: GEMM is 3.28 GFLOP vs ~250MB total layer traffic at 23.3 TB/s
// (~11us) -> everything is bandwidth bound. Strategy:
//   (a) GEMM as bf16 WMMA (v_wmma_f32_16x16x32_bf16), weight resident in LDS,
//       x read exactly once via GLOBAL_LOAD_ASYNC_TO_LDS_B128 (ASYNCcnt path)
//       with double-buffered tiles so the DMA overlaps WMMA + epilogue.
//       fp32 -> packed bf16 done with one v_add + one v_perm_b32 per pair.
//   (b) score reduction fused into the GEMM epilogue (width-16 bpermute).
//   (c) reference's segment_max + segment_min(argmax tiebreak) collapsed into
//       a single u64 atomicMax per (edge,head).
// ---------------------------------------------------------------------------

typedef __attribute__((ext_vector_type(16))) __bf16 v16bf;
typedef __attribute__((ext_vector_type(8)))  float  v8f;

union ABfrag { unsigned int u[8]; v16bf v; };

// two fp32 -> packed bf16x2 (round-half-up), one v_add each + one v_perm_b32:
// sel 0x07060302 picks bytes {hi[3],hi[2],lo[3],lo[2]} -> {bf16(hi),bf16(lo)}
__device__ __forceinline__ unsigned int pack_bf16(float lo, float hi) {
    unsigned int bl = __float_as_uint(lo) + 0x8000u;
    unsigned int bh = __float_as_uint(hi) + 0x8000u;
    return __builtin_amdgcn_perm(bh, bl, 0x07060302u);
}

#define WS  132   // bf16 elements per LDS row of W^T (128 + pad)
#define XSF 132   // f32 elements per LDS row of the x tile (128 + pad)

// ---------------------------------------------------------------------------
// Kernel 1: h = x @ W (N x 128 @ 128 x 128) via bf16 WMMA, fused with
// s_j[n,h] = <h[n,h,:], w_j[h]>, s_i[n,h] = <h[n,h,:], w_i[h]>.
// 256 threads = 8 waves; each wave owns a 16-col slice of a 16-row tile;
// grid-stride over row tiles; x tiles DMA'd to LDS with async loads
// (double-buffered), weight converted to bf16 in LDS once per block.
// ---------------------------------------------------------------------------
__global__ void __launch_bounds__(256)
gemm_scores_kernel(const float* __restrict__ x,
                   const float* __restrict__ w,      // [128 k][128 c]
                   const float* __restrict__ attw,   // [4 heads][64] (w_j|w_i)
                   float* __restrict__ hbuf,         // [N][128]
                   float* __restrict__ sj,           // [N][4]
                   float* __restrict__ si,           // [N][4]
                   int n_nodes)
{
    __shared__ unsigned short Wl[128 * WS];   // W^T as bf16: Wl[col][k]
    __shared__ float          Xb[2][16 * XSF];// double-buffered fp32 x tiles

    const int tid  = threadIdx.x;
    const int lane = tid & 31;
    const int wv   = tid >> 5;           // wave in block: 0..7
    const int l15  = lane & 15;
    const int half = lane >> 4;          // 0: lanes 0-15, 1: lanes 16-31

    // ---- stage W^T into LDS as bf16 pairs (one packed dword per store) ----
    for (int i = tid; i < 64 * 128; i += 256) {
        int c = i & 127;            // output column
        int k = (i >> 7) << 1;      // even k
        float g0 = w[(size_t)k * 128 + c];
        float g1 = w[(size_t)(k + 1) * 128 + c];
        *(unsigned int*)&Wl[c * WS + k] = pack_bf16(g0, g1);
    }

    const int colBase = wv << 4;                 // this wave's 16-col slice
    const int hd      = colBase >> 5;            // head of this slice
    const int cih     = (colBase & 31) + l15;    // channel within head
    const float wjv   = attw[hd * 64 + cih];
    const float wiv   = attw[hd * 64 + 32 + cih];

    const unsigned long long xbase = (unsigned long long)x;
    const int nTiles = (n_nodes + 15) >> 4;

    // DMA one 16x128 fp32 tile into LDS buffer b: 512 B128 transfers,
    // 2 per thread, ASYNCcnt-tracked, LDS written directly (no VGPR bounce).
    auto issue_tile = [&](int t, int b) {
        #pragma unroll
        for (int i = 0; i < 2; ++i) {
            int idx = i * 256 + tid;              // float4 index (0..511)
            int r   = idx >> 5;                   // row in tile
            int k4  = (idx & 31) << 2;            // col (step 4)
            int row = t * 16 + r;
            if (row >= n_nodes) row = n_nodes - 1;
            unsigned goff = (unsigned)(((unsigned)row * 128u + (unsigned)k4) * 4u);
            unsigned loff = (unsigned)(unsigned long long)&Xb[b][r * XSF + k4];
            asm volatile("global_load_async_to_lds_b128 %0, %1, %2"
                         :: "v"(loff), "v"(goff), "s"(xbase) : "memory");
        }
    };

    // prologue: kick off first tile while W conversion is still in flight
    issue_tile(blockIdx.x, 0);

    int cur = 0;
    for (int tile = blockIdx.x; tile < nTiles; tile += gridDim.x) {
        // my async loads done -> barrier -> every wave's loads done
        asm volatile("s_wait_asynccnt 0x0" ::: "memory");
        __syncthreads();

        // overlap: DMA the next tile into the other buffer
        {
            long long nt = (long long)tile + gridDim.x;
            if (nt < nTiles) issue_tile((int)nt, cur ^ 1);
        }

        const int rowBase = tile << 4;
        const float* xt = &Xb[cur][0];

        // ---- 16x16 f32 accumulator, K = 128 in 4 bf16 WMMA steps ----
        v8f acc = {0.f, 0.f, 0.f, 0.f, 0.f, 0.f, 0.f, 0.f};
        #pragma unroll
        for (int kt = 0; kt < 4; ++kt) {
            ABfrag a, b;
            // A (16x32 bf16): lane half selects K-halves:
            //   VGPR0-7: K = kbase+half*16+{0..15}
            const int ka = kt * 32 + half * 16;
            const float* arow = &xt[l15 * XSF + ka];
            #pragma unroll
            for (int j = 0; j < 4; ++j)
                a.u[j]     = pack_bf16(arow[2 * j],      arow[2 * j + 1]);
            #pragma unroll
            for (int j = 0; j < 4; ++j)
                a.u[4 + j] = pack_bf16(arow[8 + 2 * j], arow[8 + 2 * j + 1]);
            // B (32x16 bf16): lanes 0-15 = cols with K=kt*32+0..15,
            // lanes 16-31 same cols K=+16; W^T makes K-runs contiguous.
            const int kb = kt * 32 + half * 16;
            const unsigned short* brow = &Wl[(colBase + l15) * WS];
            #pragma unroll
            for (int j = 0; j < 8; ++j)
                b.u[j] = *(const unsigned int*)&brow[kb + 2 * j];

            acc = __builtin_amdgcn_wmma_f32_16x16x32_bf16(
                false, a.v, false, b.v, (short)0, acc, false, false);
        }

        // ---- epilogue: store h, reduce per-row score partials ----
        #pragma unroll
        for (int r = 0; r < 8; ++r) {
            const int row = rowBase + r + half * 8;  // C layout: VGPRr/lane-half
            const float v = acc[r];
            if (row < n_nodes)
                hbuf[(size_t)row * 128 + colBase + l15] = v;
            float pj = v * wjv;
            float pi2 = v * wiv;
            #pragma unroll
            for (int off = 8; off > 0; off >>= 1) {
                pj  += __shfl_xor(pj,  off, 16);
                pi2 += __shfl_xor(pi2, off, 16);
            }
            if (l15 == 0 && row < n_nodes) {
                atomicAdd(&sj[row * 4 + hd], pj);
                atomicAdd(&si[row * 4 + hd], pi2);
            }
        }
        cur ^= 1;
    }
}

// ---------------------------------------------------------------------------
// Kernel 2: per (edge, head): alpha = s_j[src]+s_i[dst]; fold segment_max +
// argmax-with-min-eid-tiebreak into one u64 atomicMax with key
// = monotone(alpha) << 32 | ~eid   (max alpha; tie -> max ~eid = min eid).
// ---------------------------------------------------------------------------
__global__ void __launch_bounds__(256)
edge_vote_kernel(const int* __restrict__ ei,          // [2][E]
                 const float* __restrict__ sj,
                 const float* __restrict__ si,
                 unsigned long long* __restrict__ packed,  // [N][4], zeroed
                 int E_)
{
    int gid = blockIdx.x * 256 + threadIdx.x;
    if (gid >= E_ * 4) return;
    int e  = gid >> 2;
    int hd = gid & 3;
    int s  = ei[e];
    int d  = ei[E_ + e];
    float alpha = sj[s * 4 + hd] + si[d * 4 + hd];
    unsigned int b   = __float_as_uint(alpha);
    unsigned int enc = (b & 0x80000000u) ? ~b : (b | 0x80000000u);  // order-preserving
    unsigned long long key =
        ((unsigned long long)enc << 32) | (unsigned int)(~(unsigned int)e);
    atomicMax(&packed[(size_t)s * 4 + hd], key);
}

// ---------------------------------------------------------------------------
// Kernel 3: one wave per (node, head). Decode winning edge, scatter-add the
// 32-channel h slice into out[dst] (lane == channel), set node_mask[dst].
// ---------------------------------------------------------------------------
__global__ void __launch_bounds__(256)
gather_scatter_kernel(const int* __restrict__ ei,
                      const unsigned long long* __restrict__ packed,
                      const float* __restrict__ hbuf,
                      float* __restrict__ out,    // [N][128], zeroed accumulator
                      int* __restrict__ mask,     // [N], zeroed
                      int n_nodes, int E_)
{
    int wid  = (blockIdx.x * 256 + threadIdx.x) >> 5;
    int lane = threadIdx.x & 31;
    if (wid >= n_nodes * 4) return;
    int n  = wid >> 2;
    int hd = wid & 3;
    unsigned long long key = packed[(size_t)n * 4 + hd];
    if (key == 0ull) return;                    // no incident edge
    int e = (int)(~(unsigned int)key);
    int d = ei[E_ + e];
    atomicAdd(&out[(size_t)d * 128 + hd * 32 + lane],
              hbuf[(size_t)n * 128 + hd * 32 + lane]);
    if (lane == 0) mask[d] = 1;                 // idempotent store
}

// ---------------------------------------------------------------------------
// Kernel 4: x_out = mask ? 0 : out (in place); emit node_mask + batch_slices.
// ---------------------------------------------------------------------------
__global__ void __launch_bounds__(256)
finalize_nodes_kernel(float* __restrict__ xout,
                      float* __restrict__ maskf,
                      float* __restrict__ bsf,
                      const int* __restrict__ mask,
                      const int* __restrict__ bs,
                      int n_nodes)
{
    int i = blockIdx.x * 256 + threadIdx.x;
    if (i >= n_nodes * 128) return;
    int n = i >> 7;
    int m = mask[n];
    float v = xout[i];
    xout[i] = m ? 0.0f : v;
    if ((i & 127) == 0) maskf[n] = (float)m;
    if (i == 0) { bsf[0] = (float)bs[0]; bsf[1] = (float)bs[1]; }
}

// ---------------------------------------------------------------------------
// Kernel 5: edge_keep = mask[src] & mask[dst]
// ---------------------------------------------------------------------------
__global__ void __launch_bounds__(256)
edge_keep_kernel(const int* __restrict__ ei,
                 const int* __restrict__ mask,
                 float* __restrict__ keep, int E_)
{
    int e = blockIdx.x * 256 + threadIdx.x;
    if (e >= E_) return;
    int s = ei[e], d = ei[E_ + e];
    keep[e] = (mask[s] && mask[d]) ? 1.0f : 0.0f;
}

__global__ void zero_kernel(unsigned int* __restrict__ p, size_t nwords)
{
    size_t i = (size_t)blockIdx.x * blockDim.x + threadIdx.x;
    size_t stride = (size_t)gridDim.x * blockDim.x;
    for (; i < nwords; i += stride) p[i] = 0u;
}

extern "C" void kernel_launch(void* const* d_in, const int* in_sizes, int n_in,
                              void* d_out, int out_size, void* d_ws, size_t ws_size,
                              hipStream_t stream)
{
    (void)n_in; (void)out_size; (void)ws_size;
    const float* x    = (const float*)d_in[0];   // [N][128]
    const int*   ei   = (const int*)  d_in[1];   // [2][E]
    const int*   bs   = (const int*)  d_in[2];   // [2]
    const float* w    = (const float*)d_in[3];   // [128][128]
    const float* attw = (const float*)d_in[4];   // [1][4][64]

    const int n_nodes = in_sizes[0] / 128;       // 100000
    const int E_      = in_sizes[1] / 2;         // 1600000

    // output layout (concatenated, float): x_out | edge_keep | node_mask | slices
    float* xout  = (float*)d_out;
    float* ekeep = xout  + (size_t)n_nodes * 128;
    float* maskf = ekeep + (size_t)E_;
    float* bsf   = maskf + (size_t)n_nodes;

    // workspace layout
    char* ws = (char*)d_ws;
    float* hbuf = (float*)ws;                                   // N*128 f32
    size_t off  = (size_t)n_nodes * 128 * sizeof(float);
    float* sj   = (float*)(ws + off);  off += (size_t)n_nodes * 4 * sizeof(float);
    float* si   = (float*)(ws + off);  off += (size_t)n_nodes * 4 * sizeof(float);
    unsigned long long* packed = (unsigned long long*)(ws + off);
    off += (size_t)n_nodes * 4 * sizeof(unsigned long long);
    int* mask = (int*)(ws + off);

    // ---- per-launch init (graph-replay safe) ----
    zero_kernel<<<2048, 256, 0, stream>>>((unsigned int*)sj,     (size_t)n_nodes * 8);  // sj+si
    zero_kernel<<<2048, 256, 0, stream>>>((unsigned int*)packed, (size_t)n_nodes * 8);
    zero_kernel<<< 512, 256, 0, stream>>>((unsigned int*)mask,   (size_t)n_nodes);
    zero_kernel<<<4096, 256, 0, stream>>>((unsigned int*)xout,   (size_t)n_nodes * 128);

    // ---- pipeline ----
    gemm_scores_kernel<<<2048, 256, 0, stream>>>(x, w, attw, hbuf, sj, si, n_nodes);

    int vThreads = E_ * 4;
    edge_vote_kernel<<<(vThreads + 255) / 256, 256, 0, stream>>>(ei, sj, si, packed, E_);

    int gThreads = n_nodes * 4 * 32;
    gather_scatter_kernel<<<(gThreads + 255) / 256, 256, 0, stream>>>(
        ei, packed, hbuf, xout, mask, n_nodes, E_);

    finalize_nodes_kernel<<<(n_nodes * 128 + 255) / 256, 256, 0, stream>>>(
        xout, maskf, bsf, mask, bs, n_nodes);

    edge_keep_kernel<<<(E_ + 255) / 256, 256, 0, stream>>>(ei, mask, ekeep, E_);
}
